// FuseBlock_27298812133916
// MI455X (gfx1250) — matrix operand order repaired and checked by reading, not compiled, not run on hardware
//
#include <hip/hip_runtime.h>

// CDNA5 / gfx1250 implementation of the FuseBlock.
// Bandwidth-bound problem (54 GFLOP vs ~1 GB traffic @ 23.3 TB/s) ->
// bf16 intermediates + V_WMMA_F32_16X16X32_BF16 for all 1x1 convs,
// LayerNorms folded algebraically into GEMM epilogues/prologues,
// FFT "attention" implemented as direct 8x8 circular convolution,
// weight tiles staged into LDS by the Tensor Data Mover (tensor_load_to_lds).

typedef unsigned short u16;
typedef unsigned int   u32;
typedef unsigned long long u64;
typedef __attribute__((ext_vector_type(16))) __bf16 v16b;
typedef __attribute__((ext_vector_type(8)))  float  v8f;
typedef __attribute__((ext_vector_type(4)))  u32    u32x4;
typedef __attribute__((ext_vector_type(8)))  u32    u32x8;

#define BATCH 2
#define IMG   256
#define HWPix 65536

__device__ __forceinline__ float bf2f(u16 h) {
  union { u32 u; float f; } c; c.u = ((u32)h) << 16; return c.f;
}
__device__ __forceinline__ u16 f2bf(float f) {
  union { float f; u32 u; } c; c.f = f;
  u32 u = c.u;
  u += 0x7fffu + ((u >> 16) & 1u);   // round-to-nearest-even
  return (u16)(u >> 16);
}

union BFrag { v16b v; u32 u[8]; };

// ---------------------------------------------------------------------------
// Per-pixel LayerNorm statistics over C channels (biased var, eps=1e-6).
// Optionally emits a bf16 copy of the fp32 input (feeds the WMMA GEMMs).
// ---------------------------------------------------------------------------
template<int C, bool BF16IN>
__global__ __launch_bounds__(256)
void ln_stats(const void* __restrict__ xin, float* __restrict__ mean,
              float* __restrict__ rstd, u16* __restrict__ copybf)
{
  const int p = blockIdx.x * 256 + threadIdx.x;
  const int b = blockIdx.y;
  const size_t base = (size_t)b * C * HWPix + p;
  float s = 0.f, s2 = 0.f;
  for (int c = 0; c < C; ++c) {
    float v;
    if (BF16IN) v = bf2f(((const u16*)xin)[base + (size_t)c * HWPix]);
    else        v = ((const float*)xin)[base + (size_t)c * HWPix];
    s += v; s2 += v * v;
    if (!BF16IN && copybf) copybf[base + (size_t)c * HWPix] = f2bf(v);
  }
  const float m = s / (float)C;
  float var = s2 / (float)C - m * m;
  var = fmaxf(var, 0.f);
  mean[(size_t)b * HWPix + p] = m;
  rstd[(size_t)b * HWPix + p] = rsqrtf(var + 1e-6f);
}

// ---------------------------------------------------------------------------
// Weight prep: W'[o,c] = bf16(w[o,c]*norm_w[c]); s[o]=rowsum(W'); t[o]=w.norm_b(+bias)
// Lets conv1x1(LN(x)) = r[p]*(W' x)[o,p] - m[p]*r[p]*s[o] + t[o].
// ---------------------------------------------------------------------------
__global__ __launch_bounds__(64)
void prep_w_ln(const float* __restrict__ w, const float* __restrict__ nw,
               const float* __restrict__ nb, const float* __restrict__ bias,
               int M, int K, u16* __restrict__ wp,
               float* __restrict__ s, float* __restrict__ t)
{
  const int o = blockIdx.x * 64 + threadIdx.x;
  if (o >= M) return;
  float ss = 0.f, tt = bias ? bias[o] : 0.f;
  for (int c = 0; c < K; ++c) {
    const u16 hb = f2bf(w[o * K + c] * nw[c]);
    wp[o * K + c] = hb;
    ss += bf2f(hb);
    tt += w[o * K + c] * nb[c];
  }
  s[o] = ss; t[o] = tt;
}

__global__ __launch_bounds__(256)
void w_to_bf16(const float* __restrict__ w, u16* __restrict__ o, int n)
{
  const int i = blockIdx.x * 256 + threadIdx.x;
  if (i < n) o[i] = f2bf(w[i]);
}

// ---------------------------------------------------------------------------
// Depthwise 3x3, SAME zero padding, bf16 in/out, fp32 weights (C,1,3,3).
// ---------------------------------------------------------------------------
__global__ __launch_bounds__(256)
void dwconv3(const u16* __restrict__ src, const float* __restrict__ w,
             u16* __restrict__ dst, int C)
{
  __shared__ u16 tile[18][20];
  const int bc = blockIdx.z;                 // b*C + c
  const int c  = bc % C;
  const int tx = threadIdx.x & 15, ty = threadIdx.x >> 4;
  const int x0 = blockIdx.x * 16, y0 = blockIdx.y * 16;
  const u16* sp = src + (size_t)bc * HWPix;
  for (int i = threadIdx.x; i < 18 * 18; i += 256) {
    const int ly = i / 18, lx = i - ly * 18;
    const int gy = y0 + ly - 1, gx = x0 + lx - 1;
    u16 v = 0;
    if (gy >= 0 && gy < IMG && gx >= 0 && gx < IMG) v = sp[gy * IMG + gx];
    tile[ly][lx] = v;
  }
  __syncthreads();
  const float* wc = w + c * 9;
  float acc = 0.f;
#pragma unroll
  for (int ky = 0; ky < 3; ++ky)
#pragma unroll
    for (int kx = 0; kx < 3; ++kx)
      acc += wc[ky * 3 + kx] * bf2f(tile[ty + ky][tx + kx]);
  dst[(size_t)bc * HWPix + (size_t)(y0 + ty) * IMG + (x0 + tx)] = f2bf(acc);
}

// ---------------------------------------------------------------------------
// "FFT attention" = per-8x8-patch circular convolution of q and k:
// attn[i,j] = sum_{u,v} q[u,v] * k[(i-u)&7, (j-v)&7].
// 8 threads per patch-channel, one output row each; q,k staged in LDS.
// ---------------------------------------------------------------------------
__global__ __launch_bounds__(256)
void fft_attn(const u16* __restrict__ qkv, u16* __restrict__ attn)
{
  __shared__ u16 qs[32][64];
  __shared__ u16 ks[32][64];
  const int g  = threadIdx.x >> 3;           // 32 patch-channels / block
  const int ti = threadIdx.x & 7;            // output row
  const int pc = blockIdx.x * 32 + g;        // patch-channel id, total B*256*1024
  const int b    = pc >> 18;                 // 256*1024 = 2^18
  const int rem  = pc & 262143;
  const int c    = rem >> 10;
  const int pidx = rem & 1023;
  const int ph = pidx >> 5, pw = pidx & 31;
  const size_t pbase = (size_t)(ph * 8) * IMG + pw * 8;
  const size_t qb = ((size_t)(b * 768 + c)) * HWPix + pbase;
  const size_t kb = ((size_t)(b * 768 + 256 + c)) * HWPix + pbase;
  *(uint4*)&qs[g][ti * 8] = *(const uint4*)(qkv + qb + (size_t)ti * IMG);
  *(uint4*)&ks[g][ti * 8] = *(const uint4*)(qkv + kb + (size_t)ti * IMG);
  __syncthreads();
  const int i = ti;
  float acc[8] = {0.f,0.f,0.f,0.f,0.f,0.f,0.f,0.f};
  for (int ku = 0; ku < 8; ++ku) {
    const int u = (i - ku) & 7;
    float qr[8], kr[8];
#pragma unroll
    for (int v = 0; v < 8; ++v) { qr[v] = bf2f(qs[g][u * 8 + v]); kr[v] = bf2f(ks[g][ku * 8 + v]); }
#pragma unroll
    for (int j = 0; j < 8; ++j)
#pragma unroll
      for (int v = 0; v < 8; ++v)
        acc[j] += qr[v] * kr[(j - v) & 7];
  }
  union { uint4 v; u16 s[8]; } out;
#pragma unroll
  for (int j = 0; j < 8; ++j) out.s[j] = f2bf(acc[j]);
  const size_t ob = ((size_t)(b * 256 + c)) * HWPix + (size_t)(ph * 8 + i) * IMG + pw * 8;
  *(uint4*)(attn + ob) = out.v;
}

// ---------------------------------------------------------------------------
// Fused WMMA GEMM: Y = A(MxK,bf16 weights) * act(KxN) with per-mode fused
// activation-build (prologue) and LN/residual epilogues.
// Tiles: 64(M) x 64(N) x 32(K); 8 waves; each wave: one 16x16 M-tile x two N-tiles.
// A tile (64 rows x 64B, row stride K*2 B) is staged into LDS by the Tensor
// Data Mover: TDM pad feature (pad 4 DWORDs every 16 DWORDs) reproduces the
// 40-u16 (80 B) padded row stride of As. One wave issues tensor_load_to_lds,
// waits on TENSORcnt, then the workgroup barrier publishes the tile.
// VGPR fragment layouts per CDNA5 ISA 7.12.2 (A per 16-bit 16x32 table;
// B in the same k order as A: lanes0-15 K=0..7,16..23 etc).
// ---------------------------------------------------------------------------
enum GemmMode { MODE_H1 = 0, MODE_P1 = 1, MODE_H2 = 2, MODE_P2 = 3 };

template<int MODE>
__global__ __launch_bounds__(256)
void gemm_ep(const u16* __restrict__ Aw, int K,
             const u16* __restrict__ act, int actC,
             const u16* __restrict__ act2, int act2C, int act2Off,
             const float* __restrict__ pmean, const float* __restrict__ prstd,
             const float* __restrict__ svec, const float* __restrict__ tvec,
             const float* __restrict__ nw2, const float* __restrict__ nb2,
             const float* __restrict__ resid, int residC,
             const float* __restrict__ scale,
             u16* __restrict__ outb, int outC,
             float* __restrict__ outf)
{
  __shared__ u16 As[64 * 40];   // [m][k], row stride 40 u16 = 64B data + 16B pad
  __shared__ u16 Bs[64 * 40];   // [n][k] (transposed on store)
  const int tid  = threadIdx.x;
  const int b    = blockIdx.z;
  const int gm0  = blockIdx.y * 64;
  const int p0   = blockIdx.x * 64;
  const int lane = tid & 31;
  const int wv   = tid >> 5;    // wave id 0..7
  const int wm   = wv & 3;
  const int wnn  = wv >> 2;     // 0..1
  const int half = lane >> 4;

  v8f acc0 = {0.f,0.f,0.f,0.f,0.f,0.f,0.f,0.f};
  v8f acc1 = {0.f,0.f,0.f,0.f,0.f,0.f,0.f,0.f};

  // B staging ownership: 4 consecutive k-rows x 2 n-columns per thread so the
  // k-contiguous Bs rows take one ds_store_b64 per column.
  const int kq = wv;            // k base = 4*kq (uniform per wave)
  const int ng = tid & 31;      // n pair = 2*ng, 2*ng+1

  // ---- TDM descriptor (all-uniform; only global_addr changes per K-step) ----
  const u32 lds_off = (u32)(size_t)(void*)As;  // flat LDS addr[31:0] == LDS offset
  u32x8 g1;
  g1[0] = (3u << 16)            // data_size = 8B units
        | (1u << 20)            // pad_enable
        | (3u << 22)            // pad_interval: 16 DWORDs (64B) between pads
        | (3u << 25);           // pad_amount: 4 DWORDs (16B)
  g1[1] = (8u << 16);           // tensor_dim0 = 8 units (= tile_dim0, no OOB)
  g1[2] = (64u << 16);          // tensor_dim1 = 64
  g1[3] = (8u << 16);           // tile_dim0 = 8 units (64B row)
  g1[4] = 64u;                  // tile_dim1 = 64 rows, tile_dim2 = 0
  g1[5] = (u32)(K >> 2);        // tensor_dim0_stride = K*2B / 8B
  g1[6] = 0u;
  g1[7] = 0u;
  u64 ga = (u64)(size_t)Aw + (u64)((size_t)gm0 * (size_t)K) * 2u;

  for (int k0 = 0; k0 < K; k0 += 32) {
    // ---- A tile via Tensor Data Mover (wave 0 only; idempotent anyway) ----
    if (wv == 0) {
      u32x4 g0;
      g0[0] = 1u;                                        // count=1
      g0[1] = lds_off;                                   // LDS dest
      g0[2] = (u32)ga;                                   // global_addr[31:0]
      g0[3] = ((u32)(ga >> 32) & 0x01FFFFFFu) | 0x80000000u; // addr[56:32] | type=2
      asm volatile("tensor_load_to_lds %0, %1" :: "s"(g0), "s"(g1) : "memory");
    }

    // ---- build + stage B (activation), transposed into Bs[n][k] ----
    const int kb = k0 + 4 * kq;
    u16 el[4], eh[4];
    if (MODE == MODE_H1 || MODE == MODE_H2) {
      const size_t ab = ((size_t)(b * actC + kb)) * HWPix + p0 + 2 * ng;
      const u32 w0 = *(const u32*)(act + ab);
      const u32 w1 = *(const u32*)(act + ab + HWPix);
      const u32 w2 = *(const u32*)(act + ab + 2 * HWPix);
      const u32 w3 = *(const u32*)(act + ab + 3 * HWPix);
      el[0] = (u16)w0; el[1] = (u16)w1; el[2] = (u16)w2; el[3] = (u16)w3;
      eh[0] = (u16)(w0 >> 16); eh[1] = (u16)(w1 >> 16);
      eh[2] = (u16)(w2 >> 16); eh[3] = (u16)(w3 >> 16);
    } else if (MODE == MODE_P1) {
      const size_t ab = ((size_t)(b * actC + kb)) * HWPix + p0 + 2 * ng;
      const size_t vb = ((size_t)(b * act2C + act2Off + kb)) * HWPix + p0 + 2 * ng;
      const int pp = b * HWPix + p0 + 2 * ng;
      const float pm0 = pmean[pp],     pr0 = prstd[pp];
      const float pm1 = pmean[pp + 1], pr1 = prstd[pp + 1];
#pragma unroll
      for (int i = 0; i < 4; ++i) {
        const u32 aw = *(const u32*)(act  + ab + (size_t)i * HWPix);
        const u32 vw = *(const u32*)(act2 + vb + (size_t)i * HWPix);
        const float wk = nw2[kb + i], bbv = nb2[kb + i];
        el[i] = f2bf(((bf2f((u16)aw)         - pm0) * pr0 * wk + bbv) * bf2f((u16)vw));
        eh[i] = f2bf(((bf2f((u16)(aw >> 16)) - pm1) * pr1 * wk + bbv) * bf2f((u16)(vw >> 16)));
      }
    } else { // MODE_P2: gated product h1*h2
      const size_t ab = ((size_t)(b * actC + kb)) * HWPix + p0 + 2 * ng;
      const size_t vb = ((size_t)(b * act2C + act2Off + kb)) * HWPix + p0 + 2 * ng;
#pragma unroll
      for (int i = 0; i < 4; ++i) {
        const u32 h1 = *(const u32*)(act  + ab + (size_t)i * HWPix);
        const u32 h2 = *(const u32*)(act2 + vb + (size_t)i * HWPix);
        el[i] = f2bf(bf2f((u16)h1) * bf2f((u16)h2));
        eh[i] = f2bf(bf2f((u16)(h1 >> 16)) * bf2f((u16)(h2 >> 16)));
      }
    }
    uint2 cl, ch;
    cl.x = (u32)el[0] | ((u32)el[1] << 16);
    cl.y = (u32)el[2] | ((u32)el[3] << 16);
    ch.x = (u32)eh[0] | ((u32)eh[1] << 16);
    ch.y = (u32)eh[2] | ((u32)eh[3] << 16);
    *(uint2*)(Bs + (2 * ng) * 40 + 4 * kq)     = cl;
    *(uint2*)(Bs + (2 * ng + 1) * 40 + 4 * kq) = ch;

    if (wv == 0) __builtin_amdgcn_s_wait_tensorcnt(0);
    __syncthreads();

    // ---- per-lane fragment gathers (contiguous 4B pairs) + WMMA ----
    BFrag a, b0, b1;
    const int mrow = wm * 16 + (lane & 15);
#pragma unroll
    for (int r = 0; r < 8; ++r) {
      const int kk = ((r >> 2) * 16) + half * 8 + ((r & 3) * 2);
      a.u[r] = *(const u32*)(As + mrow * 40 + kk);
    }
    const int nc = wnn * 32 + (lane & 15);
#pragma unroll
    for (int r = 0; r < 8; ++r) {
      const int kk = ((r >> 2) * 16) + half * 8 + ((r & 3) * 2);
      b0.u[r] = *(const u32*)(Bs + nc * 40 + kk);
      b1.u[r] = *(const u32*)(Bs + (nc + 16) * 40 + kk);
    }
    acc0 = __builtin_amdgcn_wmma_f32_16x16x32_bf16(false, a.v, false, b0.v,
                                                   (short)0, acc0, false, false);
    acc1 = __builtin_amdgcn_wmma_f32_16x16x32_bf16(false, a.v, false, b1.v,
                                                   (short)0, acc1, false, false);
    __syncthreads();
    ga += 64u;   // next 32 bf16 K-columns
  }

  // ---- epilogue: C/D layout -> m = mbase+r, n = lane&15 (+16 for 2nd tile) --
  const int mbase = gm0 + wm * 16 + half * 8;
#pragma unroll
  for (int t = 0; t < 2; ++t) {
    const v8f accv = t ? acc1 : acc0;
    const int p  = p0 + wnn * 32 + (lane & 15) + t * 16;
    const int pp = b * HWPix + p;
    if (MODE == MODE_H1 || MODE == MODE_H2) {
      const float m = pmean[pp], rs = prstd[pp];
#pragma unroll
      for (int r = 0; r < 8; ++r) {
        const int o = mbase + r;
        const float v = rs * accv[r] - m * rs * svec[o] + tvec[o];
        outb[((size_t)(b * outC + o)) * HWPix + p] = f2bf(v);
      }
    } else {
#pragma unroll
      for (int r = 0; r < 8; ++r) {
        const int o = mbase + r;
        outf[((size_t)(b * residC + o)) * HWPix + p] =
            resid[((size_t)(b * residC + o)) * HWPix + p] + accv[r] * scale[o];
      }
    }
  }
}

// ---------------------------------------------------------------------------
extern "C" void kernel_launch(void* const* d_in, const int* in_sizes, int n_in,
                              void* d_out, int out_size, void* d_ws, size_t ws_size,
                              hipStream_t stream)
{
  (void)in_sizes; (void)n_in; (void)out_size; (void)ws_size;
  const float* x    = (const float*)d_in[0];
  const float* n1w  = (const float*)d_in[1];
  const float* n1b  = (const float*)d_in[2];
  const float* w_h1 = (const float*)d_in[3];
  const float* wdw1 = (const float*)d_in[4];
  const float* n2w  = (const float*)d_in[5];
  const float* n2b  = (const float*)d_in[6];
  const float* w_p1 = (const float*)d_in[7];
  const float* n3w  = (const float*)d_in[8];
  const float* n3b  = (const float*)d_in[9];
  const float* w_h2 = (const float*)d_in[10];
  const float* b_h2 = (const float*)d_in[11];
  const float* wdw2 = (const float*)d_in[12];
  const float* w_p2 = (const float*)d_in[13];
  const float* sc1  = (const float*)d_in[14];
  const float* sc2  = (const float*)d_in[15];
  float* out = (float*)d_out;

  // Workspace carve-up (256B aligned). Assumes ws_size >= ~750 MB.
  size_t off = 0;
  auto take = [&](size_t bytes) {
    void* p = (char*)d_ws + off;
    off += (bytes + 255) & ~(size_t)255;
    return p;
  };
  const size_t HW = HWPix;
  u16*   xbf  = (u16*)  take((size_t)BATCH * 128 * HW * 2);
  float* m1   = (float*)take((size_t)BATCH * HW * 4);
  float* r1   = (float*)take((size_t)BATCH * HW * 4);
  u16*   Wp1  = (u16*)  take((size_t)768 * 128 * 2);
  float* s1   = (float*)take(768 * 4);
  float* t1   = (float*)take(768 * 4);
  u16*   hid  = (u16*)  take((size_t)BATCH * 768 * HW * 2);
  u16*   qkv  = (u16*)  take((size_t)BATCH * 768 * HW * 2);
  u16*   attn = (u16*)  take((size_t)BATCH * 256 * HW * 2);
  float* m2   = (float*)take((size_t)BATCH * HW * 4);
  float* r2   = (float*)take((size_t)BATCH * HW * 4);
  float* x1   = (float*)take((size_t)BATCH * 128 * HW * 4);
  u16*   x1bf = (u16*)  take((size_t)BATCH * 128 * HW * 2);
  float* m3   = (float*)take((size_t)BATCH * HW * 4);
  float* r3   = (float*)take((size_t)BATCH * HW * 4);
  u16*   Wp2  = (u16*)  take((size_t)256 * 128 * 2);
  float* s2   = (float*)take(256 * 4);
  float* t2   = (float*)take(256 * 4);
  u16*   wp1b = (u16*)  take((size_t)128 * 256 * 2);
  u16*   wp2b = (u16*)  take((size_t)128 * 128 * 2);
  u16*   h2b  = (u16*)  take((size_t)BATCH * 256 * HW * 2);
  u16*   dw2  = (u16*)  take((size_t)BATCH * 256 * HW * 2);

  // Weight preprocessing (tiny).
  prep_w_ln<<<dim3(12), 64, 0, stream>>>(w_h1, n1w, n1b, nullptr, 768, 128, Wp1, s1, t1);
  prep_w_ln<<<dim3(4),  64, 0, stream>>>(w_h2, n3w, n3b, b_h2,    256, 128, Wp2, s2, t2);
  w_to_bf16<<<dim3((128 * 256 + 255) / 256), 256, 0, stream>>>(w_p1, wp1b, 128 * 256);
  w_to_bf16<<<dim3((128 * 128 + 255) / 256), 256, 0, stream>>>(w_p2, wp2b, 128 * 128);

  // Branch 1: LN1 stats + bf16 copy of x.
  ln_stats<128, false><<<dim3(HWPix / 256, BATCH), 256, 0, stream>>>(x, m1, r1, xbf);
  // hidden = conv1x1(LN1(x), w_h1)  [768 ch, bf16]
  gemm_ep<MODE_H1><<<dim3(HWPix / 64, 12, BATCH), 256, 0, stream>>>(
      Wp1, 128, xbf, 128, nullptr, 0, 0, m1, r1, s1, t1,
      nullptr, nullptr, nullptr, 0, nullptr, hid, 768, nullptr);
  // qkv = dwconv3(hidden)
  dwconv3<<<dim3(IMG / 16, IMG / 16, BATCH * 768), 256, 0, stream>>>(hid, wdw1, qkv, 768);
  // attn = per-patch circular conv of q,k
  fft_attn<<<dim3((BATCH * 256 * 1024) / 32), 256, 0, stream>>>(qkv, attn);
  // LN2 stats over attn (256 ch, bf16 input)
  ln_stats<256, true><<<dim3(HWPix / 256, BATCH), 256, 0, stream>>>(attn, m2, r2, nullptr);
  // x1 = x + scale1 * conv1x1(v * LN2(attn), w_p1)   [fp32]
  gemm_ep<MODE_P1><<<dim3(HWPix / 64, 2, BATCH), 256, 0, stream>>>(
      wp1b, 256, attn, 256, qkv, 768, 512, m2, r2, nullptr, nullptr,
      n2w, n2b, x, 128, sc1, nullptr, 0, x1);

  // Branch 2 (FFN): LN3 stats + bf16 copy of x1.
  ln_stats<128, false><<<dim3(HWPix / 256, BATCH), 256, 0, stream>>>(x1, m3, r3, x1bf);
  // h = conv1x1(LN3(x1), w_h2) + b_h2  [256 ch, bf16]
  gemm_ep<MODE_H2><<<dim3(HWPix / 64, 4, BATCH), 256, 0, stream>>>(
      Wp2, 128, x1bf, 128, nullptr, 0, 0, m3, r3, s2, t2,
      nullptr, nullptr, nullptr, 0, nullptr, h2b, 256, nullptr);
  // h = dwconv3(h)
  dwconv3<<<dim3(IMG / 16, IMG / 16, BATCH * 256), 256, 0, stream>>>(h2b, wdw2, dw2, 256);
  // out = x1 + scale2 * conv1x1(h1*h2, w_p2)   [fp32 -> d_out]
  gemm_ep<MODE_P2><<<dim3(HWPix / 64, 2, BATCH), 256, 0, stream>>>(
      wp2b, 128, dw2, 256, dw2, 256, 128, nullptr, nullptr, nullptr, nullptr,
      nullptr, nullptr, x1, 128, sc2, nullptr, 0, out);
}
